// MemoryGatedAttention_61675730370619
// MI455X (gfx1250) — matrix operand order repaired and checked by reading, not compiled, not run on hardware
//
#include <hip/hip_runtime.h>
#include <hip/hip_bf16.h>
#include <math.h>

typedef __attribute__((ext_vector_type(16))) __bf16 v16bf;
typedef __attribute__((ext_vector_type(8)))  __bf16 v8bf;
typedef __attribute__((ext_vector_type(8)))  float  v8f;
typedef __attribute__((ext_vector_type(4)))  int    v4i;

// addrspace-qualified pointee typedefs for the async builtin
typedef __attribute__((address_space(1))) v4i gv4i;  // global
typedef __attribute__((address_space(3))) v4i lv4i;  // LDS

#define DEV_INLINE __device__ __forceinline__

// ---- problem constants ----
constexpr int Bq = 4, Tq = 2048, Sq = 2048, Mq = 64;
constexpr int Dq = 1024, Hq = 16, DHq = 64;
constexpr int RROWS = Bq * Tq;  // 8192

DEV_INLINE v8f wmma_bf16(v16bf a, v16bf b, v8f c) {
  return __builtin_amdgcn_wmma_f32_16x16x32_bf16(false, a, false, b, (short)0, c,
                                                 false, false);
}

// 16-byte async global->LDS copy (ASYNCcnt path); fallback: manual copy.
DEV_INLINE void async_copy16(const __bf16* g, __bf16* l) {
#if __has_builtin(__builtin_amdgcn_global_load_async_to_lds_b128)
  __builtin_amdgcn_global_load_async_to_lds_b128((gv4i*)g, (lv4i*)l, 0, 0);
#else
  *(v8bf*)l = *(const v8bf*)g;
#endif
}
DEV_INLINE void async_wait0() {
#if __has_builtin(__builtin_amdgcn_global_load_async_to_lds_b128)
  asm volatile("s_wait_asynccnt 0" ::: "memory");
#endif
}

// A-fragment (16x32 bf16, MxK) from row-major LDS tile.
// Lane L (L<16): row=L, K = {0..7,16..23}; lane L+16: row=L, K = {8..15,24..31}.
DEV_INLINE v16bf lds_a_frag(const __bf16* base, int stride, int row, int koff,
                            int lane) {
  const int kh = lane >> 4;
  const __bf16* p = base + row * stride + koff + kh * 8;
  v8bf lo = *(const v8bf*)p;
  v8bf hi = *(const v8bf*)(p + 16);
  v16bf a;
#pragma unroll
  for (int j = 0; j < 8; ++j) { a[j] = lo[j]; a[8 + j] = hi[j]; }
  return a;
}

// B-fragment (32x16 bf16, KxN) from column-major LDS (Bt[n][k]).
// Lanes 0-15: K={0..7,16..23} ; lanes 16-31: K={8..15,24..31} ; col = lane&15.
DEV_INLINE v16bf lds_b_frag(const __bf16* base, int stride, int col, int koff,
                            int lane) {
  const int kb = lane >> 4;
  const __bf16* p = base + col * stride + koff + kb * 8;
  v8bf lo = *(const v8bf*)p;
  v8bf hi = *(const v8bf*)(p + 16);
  v16bf b;
#pragma unroll
  for (int j = 0; j < 8; ++j) { b[j] = lo[j]; b[8 + j] = hi[j]; }
  return b;
}

DEV_INLINE v8bf cvt8(float4 a, float4 b) {
  v8bf v;
  v[0] = (__bf16)a.x; v[1] = (__bf16)a.y; v[2] = (__bf16)a.z; v[3] = (__bf16)a.w;
  v[4] = (__bf16)b.x; v[5] = (__bf16)b.y; v[6] = (__bf16)b.z; v[7] = (__bf16)b.w;
  return v;
}

// ------------------------------------------------------------------
// Tiled bf16-WMMA GEMM: C[8192x1024] = A[8192x1024] @ W[1024x1024] + bias
// Software-pipelined: tile k+1 global loads overlap tile k WMMAs.
// mode 0: fp32 row-major out;  mode 1: bf16 out in [B,H,T,DH] layout
// ------------------------------------------------------------------
constexpr int GBM = 128, GBN = 128, GBK = 32;
constexpr int GLDS = 40;  // padded bf16 row stride (80B, 16B-aligned)

__global__ __launch_bounds__(256) void gemm_kernel(
    const float* __restrict__ A, const float* __restrict__ W,
    const float* __restrict__ bias, float* __restrict__ outF,
    __bf16* __restrict__ outQ, int mode) {
  __shared__ __align__(16) __bf16 As[GBM * GLDS];
  __shared__ __align__(16) __bf16 Bs[GBN * GLDS];  // column-major: Bs[n][k]

  const int tid = threadIdx.x;
  const int wave = tid >> 5, lane = tid & 31;
  const int mwave = (wave & 3) * 32;   // 4 wave-rows x 32 rows
  const int nwave = (wave >> 2) * 64;  // 2 wave-cols x 64 cols
  const int row0 = blockIdx.y * GBM;
  const int col0 = blockIdx.x * GBN;

  v8f acc[2][4] = {};

  const int lrow = tid >> 1;        // A-load row
  const int lseg = (tid & 1) * 16;  // A-load 16-float segment
  const int bkr  = tid >> 3;        // B-load k-row
  const int bseg = (tid & 7) * 16;  // B-load 16-col segment

  const float* aPtr = A + (size_t)(row0 + lrow) * Dq + lseg;
  const float* wPtr = W + (size_t)bkr * Dq + col0 + bseg;

  float4 pa0, pa1, pa2, pa3;
  float pb[16];
  {  // prefetch tile 0
    const float4* ap = (const float4*)aPtr;
    pa0 = ap[0]; pa1 = ap[1]; pa2 = ap[2]; pa3 = ap[3];
#pragma unroll
    for (int j = 0; j < 16; ++j) pb[j] = wPtr[j];
  }

  for (int kt = 0; kt < Dq / GBK; ++kt) {
    // convert + stage current tile into LDS
    *(v8bf*)(As + lrow * GLDS + lseg)     = cvt8(pa0, pa1);
    *(v8bf*)(As + lrow * GLDS + lseg + 8) = cvt8(pa2, pa3);
#pragma unroll
    for (int j = 0; j < 16; ++j)
      Bs[(bseg + j) * GLDS + bkr] = (__bf16)pb[j];
    __syncthreads();

    // issue next tile's global loads; they retire during the WMMAs below
    if (kt + 1 < Dq / GBK) {
      const float4* ap = (const float4*)(aPtr + (kt + 1) * GBK);
      pa0 = ap[0]; pa1 = ap[1]; pa2 = ap[2]; pa3 = ap[3];
      const float* wp = wPtr + (size_t)(kt + 1) * GBK * Dq;
#pragma unroll
      for (int j = 0; j < 16; ++j) pb[j] = wp[j];
    }

    v16bf afr[2];
#pragma unroll
    for (int mb = 0; mb < 2; ++mb)
      afr[mb] = lds_a_frag(As, GLDS, mwave + mb * 16 + (lane & 15), 0, lane);
#pragma unroll
    for (int nb = 0; nb < 4; ++nb) {
      v16bf bfr = lds_b_frag(Bs, GLDS, nwave + nb * 16 + (lane & 15), 0, lane);
#pragma unroll
      for (int mb = 0; mb < 2; ++mb)
        acc[mb][nb] = wmma_bf16(afr[mb], bfr, acc[mb][nb]);
    }
    __syncthreads();
  }

  const int hl = lane >> 4, nl = lane & 15;
#pragma unroll
  for (int mb = 0; mb < 2; ++mb)
#pragma unroll
    for (int nb = 0; nb < 4; ++nb)
#pragma unroll
      for (int i = 0; i < 8; ++i) {
        const int r = row0 + mwave + mb * 16 + hl * 8 + i;
        const int c = col0 + nwave + nb * 16 + nl;
        const float v = acc[mb][nb][i] + bias[c];
        if (mode == 0) {
          outF[(size_t)r * Dq + c] = v;
        } else {
          const int b = r >> 11, t = r & (Tq - 1);
          const int h = c >> 6, dh = c & (DHq - 1);
          outQ[(((size_t)b * Hq + h) * Tq + t) * DHq + dh] = (__bf16)v;
        }
      }
}

// ------------------------------------------------------------------
// Flash attention per (b,h): 128 Q rows / block, S chunks of 64.
// Q/K tiles travel global->LDS over the async (ASYNCcnt) path.
// ------------------------------------------------------------------
constexpr int ATB = 128;  // T rows per block
constexpr int ASB = 64;   // S columns per chunk
constexpr int QLDS = 72;  // padded stride for 64-wide rows (144B, 16B-aligned)
constexpr int PLDS = 72;

__global__ __launch_bounds__(256) void attn_kernel(
    const __bf16* __restrict__ Q, const __bf16* __restrict__ K,
    const __bf16* __restrict__ V, const float* __restrict__ gate_p,
    float* __restrict__ out) {
  __shared__ __align__(16) __bf16 Qs[ATB * QLDS];     // 18 KB
  __shared__ __align__(16) __bf16 Ks[ASB * QLDS];     //  9 KB
  __shared__ __align__(16) __bf16 Vt[DHq * PLDS];     //  9 KB (Vt[dh][s])
  __shared__ __align__(16) __bf16 Ps[8 * 16 * PLDS];  // 18 KB (per-wave P)

  const int tid = threadIdx.x, wave = tid >> 5, lane = tid & 31;
  const int hl = lane >> 4, nl = lane & 15;
  const int bh = blockIdx.y;  // b*16 + h
  const int b = bh >> 4, h = bh & 15;
  const int t0 = blockIdx.x * ATB;

  const __bf16* qb = Q + (size_t)bh * Tq * DHq;
  const __bf16* kb = K + (size_t)bh * Sq * DHq;
  const __bf16* vb = V + (size_t)bh * Sq * DHq;

  // load Q block 128x64 via async-to-LDS
#pragma unroll
  for (int it = 0; it < 4; ++it) {
    int idx = tid + 256 * it;  // 0..1023
    int row = idx >> 3, seg = (idx & 7) * 8;
    async_copy16(qb + (size_t)(t0 + row) * DHq + seg, Qs + row * QLDS + seg);
  }
  async_wait0();
  __syncthreads();

  // Q fragments are loop-invariant: hoist them
  const __bf16* qwave = Qs + wave * 16 * QLDS;
  const v16bf aq0 = lds_a_frag(qwave, QLDS, nl, 0, lane);
  const v16bf aq1 = lds_a_frag(qwave, QLDS, nl, 32, lane);

  float mrow[8], lrow[8];
  v8f oacc[4] = {};
#pragma unroll
  for (int i = 0; i < 8; ++i) { mrow[i] = -INFINITY; lrow[i] = 0.f; }

  const float scale = gate_p[0] * 0.125f;  // gate / sqrt(DH)
  __bf16* pw = Ps + wave * 16 * PLDS;

  for (int sc = 0; sc < Sq / ASB; ++sc) {
    __syncthreads();
#pragma unroll
    for (int it = 0; it < 2; ++it) {  // K chunk 64x64 row-major, async
      int idx = tid + 256 * it;       // 0..511
      int row = idx >> 3, seg = (idx & 7) * 8;
      async_copy16(kb + (size_t)(sc * ASB + row) * DHq + seg,
                   Ks + row * QLDS + seg);
    }
#pragma unroll
    for (int it = 0; it < 8; ++it) {  // V chunk 64x64, transposed (manual)
      int idx = tid + 256 * it;       // 0..2047
      int s = idx >> 5, dp = (idx & 31) * 2;
      const __bf16* vp = vb + (size_t)(sc * ASB + s) * DHq + dp;
      Vt[dp * PLDS + s]       = vp[0];
      Vt[(dp + 1) * PLDS + s] = vp[1];
    }
    async_wait0();
    __syncthreads();

    // scores: this wave's 16-row stripe x 64 chunk columns
    v8f sacc[4];
#pragma unroll
    for (int nb = 0; nb < 4; ++nb) {
      v8f z = {};
      z = wmma_bf16(aq0, lds_b_frag(Ks, QLDS, nb * 16 + nl, 0, lane), z);
      z = wmma_bf16(aq1, lds_b_frag(Ks, QLDS, nb * 16 + nl, 32, lane), z);
      sacc[nb] = z * scale;
    }

    // online softmax (rows live in half-wave: shfl-xor masks 1,2,4,8)
#pragma unroll
    for (int i = 0; i < 8; ++i) {
      float rmax = sacc[0][i];
#pragma unroll
      for (int nb = 1; nb < 4; ++nb) rmax = fmaxf(rmax, sacc[nb][i]);
#pragma unroll
      for (int off = 1; off < 16; off <<= 1)
        rmax = fmaxf(rmax, __shfl_xor(rmax, off, 32));
      const float nm = fmaxf(mrow[i], rmax);
      const float f = __expf(mrow[i] - nm);
      mrow[i] = nm;
      float rsum = 0.f;
#pragma unroll
      for (int nb = 0; nb < 4; ++nb) {
        float p = __expf(sacc[nb][i] - nm);
        sacc[nb][i] = p;
        rsum += p;
      }
#pragma unroll
      for (int off = 1; off < 16; off <<= 1)
        rsum += __shfl_xor(rsum, off, 32);
      lrow[i] = lrow[i] * f + rsum;
#pragma unroll
      for (int nb2 = 0; nb2 < 4; ++nb2) oacc[nb2][i] *= f;
    }

    // re-layout P through per-wave LDS (LDS ops are in-order per DScnt)
#pragma unroll
    for (int nb = 0; nb < 4; ++nb)
#pragma unroll
      for (int i = 0; i < 8; ++i)
        pw[(hl * 8 + i) * PLDS + nb * 16 + nl] = (__bf16)sacc[nb][i];
    asm volatile("s_wait_dscnt 0" ::: "memory");

    // O += P(16x64) @ Vchunk(64x64)
#pragma unroll
    for (int kst = 0; kst < 2; ++kst) {
      v16bf ap = lds_a_frag(pw, PLDS, nl, kst * 32, lane);
#pragma unroll
      for (int nb2 = 0; nb2 < 4; ++nb2) {
        v16bf bv = lds_b_frag(Vt, PLDS, nb2 * 16 + nl, kst * 32, lane);
        oacc[nb2] = wmma_bf16(ap, bv, oacc[nb2]);
      }
    }
  }

  // normalize + write fp32 attention output [B,T,D]
#pragma unroll
  for (int i = 0; i < 8; ++i) {
    const float inv = 1.f / lrow[i];
    const int t = t0 + wave * 16 + hl * 8 + i;
#pragma unroll
    for (int nb2 = 0; nb2 < 4; ++nb2) {
      const int dh = nb2 * 16 + nl;
      out[((size_t)b * Tq + t) * Dq + h * DHq + dh] = oacc[nb2][i] * inv;
    }
  }
}

// ------------------------------------------------------------------
// Gate path (tiny): column means, then sigmoid-gate reduction
// ------------------------------------------------------------------
__global__ __launch_bounds__(256) void mean_kernel(const float* __restrict__ q,
                                                   const float* __restrict__ mem,
                                                   float* __restrict__ mem_in) {
  const int b = blockIdx.x;
  const int d = blockIdx.y * 256 + threadIdx.x;
  float s = 0.f;
  for (int t = 0; t < Tq; ++t) s += q[((size_t)b * Tq + t) * Dq + d];
  mem_in[b * 2 * Dq + d] = s * (1.f / Tq);
  float sm = 0.f;
  for (int m = 0; m < Mq; ++m) sm += mem[((size_t)b * Mq + m) * Dq + d];
  mem_in[b * 2 * Dq + Dq + d] = sm * (1.f / Mq);
}

__global__ __launch_bounds__(1024) void gate_kernel(
    const float* __restrict__ mem_in, const float* __restrict__ Wg,
    const float* __restrict__ bg, float* __restrict__ gate) {
  const int n = threadIdx.x;  // 1024 threads = 1024 output columns
  float acc = 0.f;
  for (int b = 0; b < Bq; ++b) {
    float dot = bg[n];
    for (int k = 0; k < 2 * Dq; ++k)
      dot += mem_in[b * 2 * Dq + k] * Wg[(size_t)k * Dq + n];
    acc += 1.f / (1.f + __expf(-dot));
  }
  __shared__ float red[1024];
  red[n] = acc;
  __syncthreads();
  for (int s = 512; s > 0; s >>= 1) {
    if (n < s) red[n] += red[n + s];
    __syncthreads();
  }
  if (n == 0) gate[0] = red[0] * (1.f / (Bq * Dq));
}

// ------------------------------------------------------------------
extern "C" void kernel_launch(void* const* d_in, const int* in_sizes, int n_in,
                              void* d_out, int out_size, void* d_ws,
                              size_t ws_size, hipStream_t stream) {
  (void)in_sizes; (void)n_in; (void)out_size; (void)ws_size;
  const float* query  = (const float*)d_in[0];
  const float* key    = (const float*)d_in[1];
  const float* value  = (const float*)d_in[2];
  const float* memory = (const float*)d_in[3];
  const float* Wq = (const float*)d_in[4];  const float* bq = (const float*)d_in[5];
  const float* Wk = (const float*)d_in[6];  const float* bk = (const float*)d_in[7];
  const float* Wv = (const float*)d_in[8];  const float* bv = (const float*)d_in[9];
  const float* Wo = (const float*)d_in[10]; const float* bo = (const float*)d_in[11];
  const float* Wg = (const float*)d_in[12]; const float* bg = (const float*)d_in[13];

  char* ws = (char*)d_ws;
  const size_t qkvBytes = (size_t)Bq * Hq * Tq * DHq * sizeof(__bf16);  // 16 MB
  __bf16* Qb = (__bf16*)(ws);
  __bf16* Kb = (__bf16*)(ws + qkvBytes);
  __bf16* Vb = (__bf16*)(ws + 2 * qkvBytes);
  float* attn  = (float*)(ws + 3 * qkvBytes);  // 32 MB fp32
  float* memin = (float*)(ws + 3 * qkvBytes + (size_t)RROWS * Dq * sizeof(float));
  float* gate  = memin + Bq * 2 * Dq;

  mean_kernel<<<dim3(Bq, Dq / 256), 256, 0, stream>>>(query, memory, memin);
  gate_kernel<<<1, 1024, 0, stream>>>(memin, Wg, bg, gate);

  dim3 ggrid(Dq / GBN, RROWS / GBM);  // (8, 64)
  gemm_kernel<<<ggrid, 256, 0, stream>>>(query, Wq, bq, nullptr, Qb, 1);
  gemm_kernel<<<ggrid, 256, 0, stream>>>(key,   Wk, bk, nullptr, Kb, 1);
  gemm_kernel<<<ggrid, 256, 0, stream>>>(value, Wv, bv, nullptr, Vb, 1);

  attn_kernel<<<dim3(Tq / ATB, Bq * Hq), 256, 0, stream>>>(Qb, Kb, Vb, gate, attn);

  gemm_kernel<<<ggrid, 256, 0, stream>>>(attn, Wo, bo, (float*)d_out, nullptr, 0);
}
